// MultiHeadAttention_81277961109773
// MI455X (gfx1250) — matrix operand order repaired and checked by reading, not compiled, not run on hardware
//
#include <hip/hip_runtime.h>
#include <hip/hip_bf16.h>

// ---------------------------------------------------------------------------
// MHA forward on gfx1250 (MI455X): bf16 WMMA GEMMs + flash-attention streaming.
// B=2, S=2048, D=1024, H=16, HD=64.
// GEMM wave tile 32x64 (A-frag pair shares B-frags); flash 32 queries/wave
// (K/V frags shared across two query sub-tiles).
// ---------------------------------------------------------------------------

typedef __attribute__((ext_vector_type(16))) __bf16          v16bf;
typedef __attribute__((ext_vector_type(8)))  float           v8f;
typedef __attribute__((ext_vector_type(16))) unsigned short  u16x16;
typedef __attribute__((ext_vector_type(8)))  unsigned short  u16x8;

static __device__ __forceinline__ unsigned short f32_to_bf16(float f) {
  unsigned int u = __builtin_bit_cast(unsigned int, f);
  unsigned int r = u + 0x7FFFu + ((u >> 16) & 1u);   // round-to-nearest-even
  return (unsigned short)(r >> 16);
}

static __device__ __forceinline__ v8f wmma_bf16(u16x16 a, u16x16 b, v8f c) {
  // D = A(16x32 bf16) * B(32x16 bf16) + C(16x16 f32)
  return __builtin_amdgcn_wmma_f32_16x16x32_bf16(
      false, __builtin_bit_cast(v16bf, a),
      false, __builtin_bit_cast(v16bf, b),
      (short)0, c, false, false);
}

// A-fragment gather: 16-bit A 16x32, contiguous k order used by this file (not the ISA table order).
// lanes 0-15 hold K 0..15, lanes 16-31 hold K 16..31 (contiguous),
// row M = lane%16.  `p` points at A[row][k_chunk_base] already offset by half*16.
static __device__ __forceinline__ u16x16 load_afrag(const unsigned short* p) {
  u16x8 c0 = *(const u16x8*)(p);
  u16x8 c1 = *(const u16x8*)(p + 8);
  u16x16 f;
#pragma unroll
  for (int t = 0; t < 8; ++t) { f[t] = c0[t]; f[t + 8] = c1[t]; }
  return f;
}

// ---------------------------------------------------------------------------
// Elementwise fp32 -> bf16
__global__ __launch_bounds__(256) void cvt_bf16(const float* __restrict__ in,
                                                unsigned short* __restrict__ out,
                                                int n) {
  int i = blockIdx.x * blockDim.x + threadIdx.x;
  if (i < n) out[i] = f32_to_bf16(in[i]);
}

// fp32 W[k][n] (1024x1024) -> bf16 Wt[n][k]
__global__ __launch_bounds__(256) void cvt_bf16_T(const float* __restrict__ in,
                                                  unsigned short* __restrict__ out) {
  int i = blockIdx.x * blockDim.x + threadIdx.x;  // linear index into out
  int n = i >> 10;          // out row  (original column)
  int k = i & 1023;         // out col  (original row)
  out[i] = f32_to_bf16(in[k * 1024 + n]);
}

// ---------------------------------------------------------------------------
// WMMA GEMM: Y[4096x1024] = A[4096x1024]bf16 * Bt[1024x1024]^T bf16 + bias
// Wave tile: 32 rows x 64 cols (2 A-fragments x 4 N-subtiles; every B-fragment
// feeds two WMMAs).  Block: 256 threads = 8 waves -> block tile 32 x 512.
// Grid: (2, 128).
//   mode 0: write bf16 to [B,H,S,HD]        (Q, K)
//   mode 1: write bf16 to [B,H,HD,S]        (V transposed)
//   mode 2: write fp32 row-major [4096x1024] (final projection)
__global__ __launch_bounds__(256) void gemm_wmma(
    const unsigned short* __restrict__ A,
    const unsigned short* __restrict__ Bt,
    const float* __restrict__ bias,
    void* __restrict__ out, int mode) {
  const int K = 1024, N = 1024;
  const int lane = threadIdx.x & 31;
  const int wave = threadIdx.x >> 5;
  const int half = lane >> 4;      // 0: lanes 0-15, 1: lanes 16-31
  const int l16  = lane & 15;
  const int m0 = blockIdx.y * 32;
  const int n0 = blockIdx.x * 512 + wave * 64;

  const unsigned short* arow0 = A + (size_t)(m0 + l16) * K + half * 16;
  const unsigned short* arow1 = A + (size_t)(m0 + 16 + l16) * K + half * 16;
  v8f acc[2][4] = {};

  for (int kc = 0; kc < K; kc += 32) {
    u16x16 af0 = load_afrag(arow0 + kc);
    u16x16 af1 = load_afrag(arow1 + kc);
    // Hint the streamed activation rows into cache ahead of use.
    __builtin_prefetch(arow0 + kc + 256, 0, 0);
    __builtin_prefetch(arow1 + kc + 256, 0, 0);

#pragma unroll
    for (int ns = 0; ns < 4; ++ns) {
      // B fragment: lane holds 16 consecutive K values of its column.
      const unsigned short* bp =
          Bt + (size_t)(n0 + ns * 16 + l16) * K + kc + half * 16;
      u16x16 bf = *(const u16x16*)bp;
      acc[0][ns] = wmma_bf16(af0, bf, acc[0][ns]);
      acc[1][ns] = wmma_bf16(af1, bf, acc[1][ns]);
    }
  }

  // Epilogue. C/D layout: VGPR r holds row (half?8+r:r), column = lane%16.
#pragma unroll
  for (int ms = 0; ms < 2; ++ms) {
#pragma unroll
    for (int ns = 0; ns < 4; ++ns) {
      int col = n0 + ns * 16 + l16;
      float bcol = bias[col];
#pragma unroll
      for (int r = 0; r < 8; ++r) {
        int row = m0 + ms * 16 + half * 8 + r;
        float v = acc[ms][ns][r] + bcol;
        if (mode == 2) {
          ((float*)out)[(size_t)row * N + col] = v;
        } else {
          int b = row >> 11, s = row & 2047;     // row = b*2048 + s
          int h = col >> 6,  hd = col & 63;      // col = h*64 + hd
          size_t idx = (mode == 0)
              ? (((size_t)(b * 16 + h) * 2048 + s) * 64 + hd)
              : (((size_t)(b * 16 + h) * 64 + hd) * 2048 + s);
          ((unsigned short*)out)[idx] = f32_to_bf16(v);
        }
      }
    }
  }
}

// ---------------------------------------------------------------------------
// Flash attention: one wave per (b,h, 32-query tile). Streams 32 keys/step;
// every K/V fragment is reused by two 16-query sub-tiles.
// Q,K: bf16 [BH, S, 64].  Vt: bf16 [BH, 64, S].  Ob: bf16 [B, S, 1024].
__global__ __launch_bounds__(256) void flash_wmma(
    const unsigned short* __restrict__ Qb,
    const unsigned short* __restrict__ Kb,
    const unsigned short* __restrict__ Vt,
    unsigned short* __restrict__ Ob) {
  const int S = 2048;
  const int lane = threadIdx.x & 31;
  const int wave = threadIdx.x >> 5;
  const int gw = blockIdx.x * 8 + wave;   // 0..2047
  const int bh = gw >> 6;                 // 0..31  (b*16 + h)
  const int qt = gw & 63;                 // 32-query tile within sequence
  const int half = lane >> 4;
  const int l16  = lane & 15;

  __shared__ __align__(64) unsigned short pT[8][2 * 16 * 32]; // per-wave P tiles
  unsigned short* pW = pT[wave];

  // Q A-fragments: 2 query sub-tiles x 2 K-chunks (HD=64).
  u16x16 aq[2][2];
#pragma unroll
  for (int qs = 0; qs < 2; ++qs) {
    const unsigned short* qbase =
        Qb + ((size_t)bh * S + qt * 32 + qs * 16) * 64;
#pragma unroll
    for (int kc = 0; kc < 2; ++kc)
      aq[qs][kc] = load_afrag(qbase + (size_t)l16 * 64 + kc * 32 + half * 16);
  }

  v8f o[2][4] = {};
  float m[2][8], l[2][8];
#pragma unroll
  for (int qs = 0; qs < 2; ++qs)
#pragma unroll
    for (int r = 0; r < 8; ++r) { m[qs][r] = -3.0e38f; l[qs][r] = 0.f; }

  const unsigned short* kbase = Kb + (size_t)bh * S * 64;
  const unsigned short* vbase = Vt + (size_t)bh * 64 * S;

  for (int kb = 0; kb < S; kb += 32) {
    // --- K B-fragments (shared by both query sub-tiles) ---
    u16x16 bk[2][2];  // [key sub-tile][k-chunk]
#pragma unroll
    for (int ks = 0; ks < 2; ++ks)
#pragma unroll
      for (int kc = 0; kc < 2; ++kc)
        bk[ks][kc] = *(const u16x16*)(kbase + (size_t)(kb + ks * 16 + l16) * 64 +
                                      kc * 32 + half * 16);

#pragma unroll
    for (int qs = 0; qs < 2; ++qs) {
      // --- scores ---
      v8f s0 = {}, s1 = {};
#pragma unroll
      for (int kc = 0; kc < 2; ++kc) {
        s0 = wmma_bf16(aq[qs][kc], bk[0][kc], s0);
        s1 = wmma_bf16(aq[qs][kc], bk[1][kc], s1);
      }

      // --- online softmax (per row; rows live in 16-lane halves) ---
      float fac[8];
#pragma unroll
      for (int r = 0; r < 8; ++r) {
        float a = s0[r] * 0.125f;          // 1/sqrt(64)
        float b2 = s1[r] * 0.125f;
        float mx = fmaxf(a, b2);
        mx = fmaxf(mx, __shfl_xor(mx, 1, 32));
        mx = fmaxf(mx, __shfl_xor(mx, 2, 32));
        mx = fmaxf(mx, __shfl_xor(mx, 4, 32));
        mx = fmaxf(mx, __shfl_xor(mx, 8, 32));
        float nm = fmaxf(m[qs][r], mx);
        fac[r] = __expf(m[qs][r] - nm);
        m[qs][r] = nm;
        float p0 = __expf(a - nm);
        float p1 = __expf(b2 - nm);
        float ps = p0 + p1;
        ps += __shfl_xor(ps, 1, 32);
        ps += __shfl_xor(ps, 2, 32);
        ps += __shfl_xor(ps, 4, 32);
        ps += __shfl_xor(ps, 8, 32);
        l[qs][r] = l[qs][r] * fac[r] + ps;
        int row = half * 8 + r;            // C/D row this VGPR belongs to
        pW[qs * 512 + row * 32 + l16]      = f32_to_bf16(p0);
        pW[qs * 512 + row * 32 + 16 + l16] = f32_to_bf16(p1);
      }
#pragma unroll
      for (int n = 0; n < 4; ++n)
#pragma unroll
        for (int r = 0; r < 8; ++r) o[qs][n][r] *= fac[r];
    }

    // Ensure P tiles are resident in LDS before cross-lane A-fragment reads.
    asm volatile("s_wait_dscnt 0x0" ::: "memory");

    // --- P A-fragments (16x32 each) from LDS ---
    u16x16 ap[2];
#pragma unroll
    for (int qs = 0; qs < 2; ++qs)
      ap[qs] = load_afrag(pW + qs * 512 + l16 * 32 + half * 16);

    // --- context: O += P * V; each V fragment feeds both query sub-tiles ---
#pragma unroll
    for (int n = 0; n < 4; ++n) {
      u16x16 bv = *(const u16x16*)(vbase + (size_t)(n * 16 + l16) * S +
                                   kb + half * 16);
      o[0][n] = wmma_bf16(ap[0], bv, o[0][n]);
      o[1][n] = wmma_bf16(ap[1], bv, o[1][n]);
    }
  }

  // Normalize and store bf16 to [B, S, 1024] (concat-heads layout).
  int b = bh >> 4, h = bh & 15;
#pragma unroll
  for (int qs = 0; qs < 2; ++qs) {
    unsigned short* orow =
        Ob + ((size_t)(b * 2048 + qt * 32 + qs * 16) * 1024) + h * 64;
#pragma unroll
    for (int n = 0; n < 4; ++n) {
      int col = n * 16 + l16;
#pragma unroll
      for (int r = 0; r < 8; ++r) {
        int row = half * 8 + r;
        float inv = 1.0f / l[qs][r];
        orow[(size_t)row * 1024 + col] = f32_to_bf16(o[qs][n][r] * inv);
      }
    }
  }
}

// ---------------------------------------------------------------------------
extern "C" void kernel_launch(void* const* d_in, const int* in_sizes, int n_in,
                              void* d_out, int out_size, void* d_ws, size_t ws_size,
                              hipStream_t stream) {
  (void)in_sizes; (void)n_in; (void)out_size; (void)ws_size;
  const float* x  = (const float*)d_in[0];
  const float* Wq = (const float*)d_in[1];
  const float* bq = (const float*)d_in[2];
  const float* Wk = (const float*)d_in[3];
  const float* bk = (const float*)d_in[4];
  const float* Wv = (const float*)d_in[5];
  const float* bv = (const float*)d_in[6];
  const float* Wo = (const float*)d_in[7];
  const float* bo = (const float*)d_in[8];

  char* ws = (char*)d_ws;
  size_t off = 0;
  auto carve = [&](size_t bytes) -> unsigned short* {
    unsigned short* p = (unsigned short*)(ws + off);
    off += (bytes + 255) & ~(size_t)255;
    return p;
  };
  const size_t MK = 4096ull * 1024ull;         // x / O / Q / K / V element count
  unsigned short* xb  = carve(MK * 2);         // bf16 x            (8 MB)
  unsigned short* wqt = carve(1024ull*1024*2); // bf16 Wq^T         (2 MB)
  unsigned short* wkt = carve(1024ull*1024*2);
  unsigned short* wvt = carve(1024ull*1024*2);
  unsigned short* wot = carve(1024ull*1024*2);
  unsigned short* Qb  = carve(MK * 2);         // [BH,S,64]
  unsigned short* Kb  = carve(MK * 2);         // [BH,S,64]
  unsigned short* Vtb = carve(MK * 2);         // [BH,64,S]
  unsigned short* Ob  = carve(MK * 2);         // [B,S,1024]

  // 1) precision conversion (+ weight transpose for contiguous B-fragments)
  cvt_bf16<<<(int)(MK / 256), 256, 0, stream>>>(x, xb, (int)MK);
  cvt_bf16_T<<<1024 * 1024 / 256, 256, 0, stream>>>(Wq, wqt);
  cvt_bf16_T<<<1024 * 1024 / 256, 256, 0, stream>>>(Wk, wkt);
  cvt_bf16_T<<<1024 * 1024 / 256, 256, 0, stream>>>(Wv, wvt);
  cvt_bf16_T<<<1024 * 1024 / 256, 256, 0, stream>>>(Wo, wot);

  // 2) QKV projections (WMMA), writing attention-friendly layouts
  dim3 ggrid(2, 128);
  gemm_wmma<<<ggrid, 256, 0, stream>>>(xb, wqt, bq, (void*)Qb,  0);
  gemm_wmma<<<ggrid, 256, 0, stream>>>(xb, wkt, bk, (void*)Kb,  0);
  gemm_wmma<<<ggrid, 256, 0, stream>>>(xb, wvt, bv, (void*)Vtb, 1);

  // 3) flash attention (scores never hit HBM)
  flash_wmma<<<256, 256, 0, stream>>>(Qb, Kb, Vtb, Ob);

  // 4) output projection -> fp32 d_out
  gemm_wmma<<<ggrid, 256, 0, stream>>>(Ob, wot, bo, d_out, 2);
}
